// MTModel_74302934220932
// MI455X (gfx1250) — matrix operand order repaired and checked by reading, not compiled, not run on hardware
//
#include <hip/hip_runtime.h>
#include <hip/hip_bf16.h>

// ---------------------------------------------------------------------------
// MTModel fused forward for gfx1250 (MI455X). T=13, B=16, S=512, D=768, L=1.
// Roofline: 3 x 326MB f32 hidden states read once = ~980MB -> ~42us at
// 23.3 TB/s. All GEMMs total ~8 GFLOP (M=16 skinny) and all weights (~77MB)
// are L2-resident (192MB L2), so they are off the critical path. Structure:
//  - flash-style single pass over m1_other (LN folded into logits
//    algebraically, online softmax, one HBM read of hs)
//  - skinny GEMMs as 16x16 WMMA tiles (M = batch = 16)
//  - 5 fused launches per scan step to keep graph overhead << 42us.
// ---------------------------------------------------------------------------

typedef __attribute__((ext_vector_type(16))) __bf16 v16bf;
typedef __attribute__((ext_vector_type(8)))  float  v8f;
typedef __attribute__((ext_vector_type(2)))  float  v2f;

// Prefer exact f32 WMMA (V_WMMA_F32_16X16X4_F32); fall back to the
// codegen-proven bf16 16x16x32 path if the builtin is not declared.
#if __has_builtin(__builtin_amdgcn_wmma_f32_16x16x4_f32)
#define USE_F32_WMMA 1
#else
#define USE_F32_WMMA 0
#endif

#define DD   768
#define BB   16
#define SS   512
#define BDF  12288          // B*D floats, one ws buffer unit
#define NW   16             // waves per attention block

// ws buffer indices (each BDF floats, per modality):
//  0..2 curr | 3..5 lncls | 6..8 lncurr | 9..11 map1pre | 12..14 map1 | 15..17 map2

__device__ __forceinline__ float wred32(float v) {
  #pragma unroll
  for (int o = 16; o > 0; o >>= 1) v += __shfl_xor(v, o, 32);
  return v;
}

__device__ __forceinline__ float sigmoidf(float x) {
  return 1.0f / (1.0f + __expf(-x));
}

// Row LayerNorm over 768 elements, blockDim=256 helper.
__device__ __forceinline__ void ln_row_768(const float* __restrict__ in,
                                           float* __restrict__ out,
                                           float* s1, float* s2) {
  const int tid = threadIdx.x;
  float x0 = in[tid], x1 = in[tid + 256], x2 = in[tid + 512];
  s1[tid] = x0 + x1 + x2;
  s2[tid] = x0 * x0 + x1 * x1 + x2 * x2;
  __syncthreads();
  for (int o = 128; o > 0; o >>= 1) {
    if (tid < o) { s1[tid] += s1[tid + o]; s2[tid] += s2[tid + o]; }
    __syncthreads();
  }
  const float mean = s1[0] * (1.0f / 768.0f);
  const float var  = s2[0] * (1.0f / 768.0f) - mean * mean;
  const float rstd = rsqrtf(var + 1e-5f);
  out[tid]       = (x0 - mean) * rstd;
  out[tid + 256] = (x1 - mean) * rstd;
  out[tid + 512] = (x2 - mean) * rstd;
}

// ---------------------------------------------------------------------------
// Stage 1: LN of cls tokens hs[t,:,0,:] and of the carry, all 3 modalities.
// grid = 96 (3 mod x {16 cls rows, 16 curr rows}), blockDim = 256.
// ---------------------------------------------------------------------------
__global__ void __launch_bounds__(256)
ln_stage1_k(const float* __restrict__ hs0, const float* __restrict__ hs1,
            const float* __restrict__ hs2, unsigned long long tOff,
            float* __restrict__ ws) {
  __shared__ float s1[256], s2[256];
  const int m   = blockIdx.x / 32;
  const int sub = blockIdx.x % 32;
  const float* hs = (m == 0) ? hs0 : (m == 1 ? hs1 : hs2);
  const float* in;
  float* out;
  if (sub < 16) {          // cls token row b=sub (s = 0)
    in  = hs + tOff + (size_t)sub * SS * DD;
    out = ws + (size_t)(3 + m) * BDF + sub * DD;
  } else {                 // carry row
    const int b = sub - 16;
    in  = ws + (size_t)m * BDF + b * DD;
    out = ws + (size_t)(6 + m) * BDF + b * DD;
  }
  ln_row_768(in, out, s1, s2);
}

// Stage 3: map1 = LN(map1pre). grid = 48 (3 mod x 16 rows).
__global__ void __launch_bounds__(256)
ln_stage2_k(float* __restrict__ ws) {
  __shared__ float s1[256], s2[256];
  const int m = blockIdx.x / 16, b = blockIdx.x % 16;
  ln_row_768(ws + (size_t)(9 + m) * BDF + b * DD,
             ws + (size_t)(12 + m) * BDF + b * DD, s1, s2);
}

// ---------------------------------------------------------------------------
// Stage 2: map1pre = [lnc2|lnc3] (16x1536) @ Wm1 + bm1, all modalities.
// One wave per 16x16 tile; grid = 144 (3 mod x 48 tiles).
// Fragment layouts per CDNA5 ISA §7.12.2.
// ---------------------------------------------------------------------------
__global__ void __launch_bounds__(32)
gemm_map1_k(float* __restrict__ ws,
            const float* __restrict__ W0, const float* __restrict__ W1,
            const float* __restrict__ W2, const float* __restrict__ c0,
            const float* __restrict__ c1, const float* __restrict__ c2b) {
  const int mod  = blockIdx.x / 48;
  const int tile = blockIdx.x % 48;
  const int lane = threadIdx.x;
  const int m    = lane & 15;
  const int hi   = lane >> 4;
  const int n    = tile * 16 + (lane & 15);
  const int c2i[3] = {1, 0, 0}, c3i[3] = {2, 2, 1};
  const float* lnc2 = ws + (size_t)(3 + c2i[mod]) * BDF;
  const float* lnc3 = ws + (size_t)(3 + c3i[mod]) * BDF;
  const float* W    = (mod == 0) ? W0 : (mod == 1 ? W1 : W2);
  const float* bias = (mod == 0) ? c0 : (mod == 1 ? c1 : c2b);
  float* out = ws + (size_t)(9 + mod) * BDF;

  v8f acc = {};
#if USE_F32_WMMA
  // A 16x4 f32: VGPR0 = K0(lanes0-15)/K2(16-31), VGPR1 = K1/K3. B mirrored.
  for (int kb = 0; kb < 1536; kb += 4) {
    const int ka = kb + hi * 2;
    v2f a, b;
    a[0] = (ka     < 768) ? lnc2[m * DD + ka]     : lnc3[m * DD + ka - 768];
    a[1] = (ka + 1 < 768) ? lnc2[m * DD + ka + 1] : lnc3[m * DD + ka + 1 - 768];
    b[0] = W[(size_t)ka * DD + n];
    b[1] = W[(size_t)(ka + 1) * DD + n];
    acc = __builtin_amdgcn_wmma_f32_16x16x4_f32(false, a, false, b,
                                                (short)0, acc, false, false);
  }
#else
  for (int kb = 0; kb < 1536; kb += 32) {
    v16bf a, b;
    #pragma unroll
    for (int v = 0; v < 8; ++v) {
      int ka = kb + ((v < 4) ? (2 * v + hi * 8) : (16 + 2 * (v - 4) + hi * 8));
      float a0 = (ka     < 768) ? lnc2[m * DD + ka]     : lnc3[m * DD + ka - 768];
      float a1 = (ka + 1 < 768) ? lnc2[m * DD + ka + 1] : lnc3[m * DD + ka + 1 - 768];
      a[2 * v]     = (__bf16)a0;
      a[2 * v + 1] = (__bf16)a1;
      int kr = ka;
      b[2 * v]     = (__bf16)W[(size_t)kr * DD + n];
      b[2 * v + 1] = (__bf16)W[(size_t)(kr + 1) * DD + n];
    }
    acc = __builtin_amdgcn_wmma_f32_16x16x32_bf16(false, a, false, b,
                                                  (short)0, acc, false, false);
  }
#endif
  #pragma unroll
  for (int r = 0; r < 8; ++r) {
    int mr = r + hi * 8;
    out[mr * DD + n] = acc[r] + bias[n];
  }
}

// ---------------------------------------------------------------------------
// Stage 4: gates[g] = sigmoid(cat3 @ Wm2[g] + bm2[g]); map2 = sum g_i * x_i.
// cat3 = [map1|lncurr|lncls] (16x2304). One wave / 16x16 tile, 3 accumulators.
// grid = 144.
// ---------------------------------------------------------------------------
__device__ __forceinline__ float cat3_at(const float* m1, const float* lc,
                                         const float* cl, int m, int k) {
  if (k < 768)  return m1[m * DD + k];
  if (k < 1536) return lc[m * DD + (k - 768)];
  return cl[m * DD + (k - 1536)];
}

__global__ void __launch_bounds__(32)
gemm_gates_k(float* __restrict__ ws,
             const float* __restrict__ Wa, const float* __restrict__ Wb,
             const float* __restrict__ Wc, const float* __restrict__ ba,
             const float* __restrict__ bbv, const float* __restrict__ bc) {
  const int mod  = blockIdx.x / 48;
  const int tile = blockIdx.x % 48;
  const int lane = threadIdx.x;
  const int m    = lane & 15;
  const int hi   = lane >> 4;
  const int n    = tile * 16 + (lane & 15);
  const float* map1   = ws + (size_t)(12 + mod) * BDF;
  const float* lncurr = ws + (size_t)(6 + mod) * BDF;
  const float* lncls  = ws + (size_t)(3 + mod) * BDF;
  const float* Wm2    = (mod == 0) ? Wa : (mod == 1 ? Wb : Wc);
  const float* bm2    = (mod == 0) ? ba : (mod == 1 ? bbv : bc);
  float* map2 = ws + (size_t)(15 + mod) * BDF;
  const float* W0 = Wm2;
  const float* W1 = Wm2 + (size_t)2304 * DD;
  const float* W2 = Wm2 + (size_t)2 * 2304 * DD;

  v8f acc0 = {}, acc1 = {}, acc2 = {};
#if USE_F32_WMMA
  for (int kb = 0; kb < 2304; kb += 4) {
    const int ka = kb + hi * 2;
    v2f a, b0, b1, b2;
    a[0] = cat3_at(map1, lncurr, lncls, m, ka);
    a[1] = cat3_at(map1, lncurr, lncls, m, ka + 1);
    const size_t o0 = (size_t)ka * DD + n, o1 = (size_t)(ka + 1) * DD + n;
    b0[0] = W0[o0]; b0[1] = W0[o1];
    b1[0] = W1[o0]; b1[1] = W1[o1];
    b2[0] = W2[o0]; b2[1] = W2[o1];
    acc0 = __builtin_amdgcn_wmma_f32_16x16x4_f32(false, a, false, b0, (short)0, acc0, false, false);
    acc1 = __builtin_amdgcn_wmma_f32_16x16x4_f32(false, a, false, b1, (short)0, acc1, false, false);
    acc2 = __builtin_amdgcn_wmma_f32_16x16x4_f32(false, a, false, b2, (short)0, acc2, false, false);
  }
#else
  for (int kb = 0; kb < 2304; kb += 32) {
    v16bf a, b0, b1, b2;
    #pragma unroll
    for (int v = 0; v < 8; ++v) {
      int ka = kb + ((v < 4) ? (2 * v + hi * 8) : (16 + 2 * (v - 4) + hi * 8));
      a[2 * v]     = (__bf16)cat3_at(map1, lncurr, lncls, m, ka);
      a[2 * v + 1] = (__bf16)cat3_at(map1, lncurr, lncls, m, ka + 1);
      int kr = ka;
      size_t o0 = (size_t)kr * DD + n, o1 = (size_t)(kr + 1) * DD + n;
      b0[2 * v] = (__bf16)W0[o0]; b0[2 * v + 1] = (__bf16)W0[o1];
      b1[2 * v] = (__bf16)W1[o0]; b1[2 * v + 1] = (__bf16)W1[o1];
      b2[2 * v] = (__bf16)W2[o0]; b2[2 * v + 1] = (__bf16)W2[o1];
    }
    acc0 = __builtin_amdgcn_wmma_f32_16x16x32_bf16(false, a, false, b0, (short)0, acc0, false, false);
    acc1 = __builtin_amdgcn_wmma_f32_16x16x32_bf16(false, a, false, b1, (short)0, acc1, false, false);
    acc2 = __builtin_amdgcn_wmma_f32_16x16x32_bf16(false, a, false, b2, (short)0, acc2, false, false);
  }
#endif
  #pragma unroll
  for (int r = 0; r < 8; ++r) {
    int mr = r + hi * 8;
    float g0 = sigmoidf(acc0[r] + bm2[n]);
    float g1 = sigmoidf(acc1[r] + bm2[768 + n]);
    float g2 = sigmoidf(acc2[r] + bm2[1536 + n]);
    map2[mr * DD + n] = g0 * map1[mr * DD + n] + g1 * lncurr[mr * DD + n]
                      + g2 * lncls[mr * DD + n];
  }
}

// ---------------------------------------------------------------------------
// Stage 5: fused attention + carry update, all modalities.
// grid = 48 (mod*16 + b), blockDim = 512 (16 waves). Per wave: online
// softmax over its share of rows s=1..511 with per-lane acc[24]; explicit
// prefetch of the row NW ahead (stride 96B touches every 128B line).
// ---------------------------------------------------------------------------
__global__ void __launch_bounds__(512)
attn_k(const float* __restrict__ hs0, const float* __restrict__ hs1,
       const float* __restrict__ hs2, unsigned long long tOff,
       const float* __restrict__ wa0, const float* __restrict__ wa1,
       const float* __restrict__ wa2, const float* __restrict__ ba0,
       const float* __restrict__ ba1, const float* __restrict__ ba2,
       float* __restrict__ ws) {
  __shared__ float accbuf[NW][768];
  __shared__ float redM[NW], redZs[NW];
  __shared__ float s1[512], s2[512];
  const int mod = blockIdx.x >> 4;
  const int b   = blockIdx.x & 15;
  const int tid = threadIdx.x;
  const int wid = tid >> 5, lane = tid & 31;
  const float* hs   = (mod == 0) ? hs0 : (mod == 1 ? hs1 : hs2);
  const float* watt = (mod == 0) ? wa0 : (mod == 1 ? wa1 : wa2);
  const float* batt = (mod == 0) ? ba0 : (mod == 1 ? ba1 : ba2);
  const float* map2 = ws + (size_t)(15 + mod) * BDF + b * DD;
  float* curr       = ws + (size_t)mod * BDF + b * DD;

  // bias_b = map2[b,:] . watt[768:1536] + batt
  float p = 0.0f;
  for (int d = tid; d < 768; d += 512) p += map2[d] * watt[768 + d];
  s1[tid] = p; __syncthreads();
  for (int o = 256; o > 0; o >>= 1) {
    if (tid < o) s1[tid] += s1[tid + o];
    __syncthreads();
  }
  const float bias = s1[0] + batt[0];
  __syncthreads();

  // per-lane watt[:768] slice; d(lane,j) = lane*4 + 128*(j/4) + (j%4)
  float wv[24]; float swp = 0.0f;
  #pragma unroll
  for (int q = 0; q < 6; ++q) {
    float4 w4 = ((const float4*)watt)[lane + 32 * q];
    wv[4*q] = w4.x; wv[4*q+1] = w4.y; wv[4*q+2] = w4.z; wv[4*q+3] = w4.w;
    swp += w4.x + w4.y + w4.z + w4.w;
  }
  const float SW = wred32(swp);

  float acc[24];
  #pragma unroll
  for (int j = 0; j < 24; ++j) acc[j] = 0.0f;
  float mw = -1e30f, Zw = 0.0f;
  const float* rows0 = hs + tOff + (size_t)b * SS * DD;
  const float4* base = (const float4*)rows0;
  const char*  pbase = (const char*)rows0;
  for (int s = 1 + wid; s < SS; s += NW) {
    if (s + NW < SS)  // prefetch next row handled by this wave
      __builtin_prefetch((const void*)(pbase + (size_t)(s + NW) * 3072 + lane * 96), 0, 3);
    const float4* row = base + (size_t)s * 192;
    float x[24]; float sx = 0, sxx = 0, sxw = 0;
    #pragma unroll
    for (int q = 0; q < 6; ++q) {
      float4 v = row[lane + 32 * q];
      x[4*q] = v.x; x[4*q+1] = v.y; x[4*q+2] = v.z; x[4*q+3] = v.w;
    }
    #pragma unroll
    for (int j = 0; j < 24; ++j) { sx += x[j]; sxx += x[j]*x[j]; sxw += x[j]*wv[j]; }
    sx = wred32(sx); sxx = wred32(sxx); sxw = wred32(sxw);
    const float mean = sx * (1.0f / 768.0f);
    const float var  = sxx * (1.0f / 768.0f) - mean * mean;
    const float rstd = rsqrtf(var + 1e-5f);
    const float logit = (sxw - mean * SW) * rstd + bias;   // ln(x).watt + bias
    const float mnew  = fmaxf(mw, logit);
    const float alpha = __expf(mw - mnew);
    const float w     = __expf(logit - mnew);
    Zw = Zw * alpha + w;
    const float wrs = w * rstd;
    #pragma unroll
    for (int j = 0; j < 24; ++j) acc[j] = acc[j] * alpha + wrs * (x[j] - mean);
    mw = mnew;
  }
  if (lane == 0) redM[wid] = mw;
  __syncthreads();
  float M = redM[0];
  #pragma unroll
  for (int i = 1; i < NW; ++i) M = fmaxf(M, redM[i]);
  const float e = __expf(mw - M);          // wave-uniform
  if (lane == 0) redZs[wid] = Zw * e;
  #pragma unroll
  for (int j = 0; j < 24; ++j)
    accbuf[wid][lane * 4 + (j >> 2) * 128 + (j & 3)] = acc[j] * e;
  __syncthreads();
  float Zt = 0.0f;
  #pragma unroll
  for (int i = 0; i < NW; ++i) Zt += redZs[i];
  const float inv = 1.0f / Zt;

  // new_t + map2, then row LN -> carry. d = tid and (tid<256 ? tid+512 : -)
  float y0, y1 = 0.0f; float ps = 0, pss = 0;
  {
    float a = 0.0f;
    #pragma unroll
    for (int w2 = 0; w2 < NW; ++w2) a += accbuf[w2][tid];
    y0 = a * inv + map2[tid]; ps += y0; pss += y0 * y0;
    if (tid < 256) {
      const int d = tid + 512;
      a = 0.0f;
      #pragma unroll
      for (int w2 = 0; w2 < NW; ++w2) a += accbuf[w2][d];
      y1 = a * inv + map2[d]; ps += y1; pss += y1 * y1;
    }
  }
  s1[tid] = ps; s2[tid] = pss; __syncthreads();
  for (int o = 256; o > 0; o >>= 1) {
    if (tid < o) { s1[tid] += s1[tid + o]; s2[tid] += s2[tid + o]; }
    __syncthreads();
  }
  const float mean = s1[0] * (1.0f / 768.0f);
  const float var  = s2[0] * (1.0f / 768.0f) - mean * mean;
  const float rstd = rsqrtf(var + 1e-5f);
  curr[tid] = (y0 - mean) * rstd;
  if (tid < 256) curr[tid + 512] = (y1 - mean) * rstd;
}

// ---------------------------------------------------------------------------
// Final heads: out[b] = Wwf.[t.Wp0+bp0, v.Wp1+bp1, a.Wp2+bp2, cat(t,v,a).Wf+bf]
// ---------------------------------------------------------------------------
__global__ void __launch_bounds__(256)
final_k(const float* __restrict__ ws, const float* __restrict__ Wp,
        const float* __restrict__ bp, const float* __restrict__ Wf,
        const float* __restrict__ bf, const float* __restrict__ Wwf,
        float* __restrict__ out) {
  __shared__ float s[4][256];
  const int b = blockIdx.x, tid = threadIdx.x;
  const float* te = ws + 0 * BDF;   // curr[text]
  const float* ae = ws + 1 * BDF;   // curr[audio]
  const float* ve = ws + 2 * BDF;   // curr[video]
  float p0 = 0, p1 = 0, p2 = 0, pm = 0;
  for (int d = tid; d < 768; d += 256) {
    const float tv = te[b * DD + d], av = ae[b * DD + d], vv = ve[b * DD + d];
    p0 += tv * Wp[d];
    p1 += vv * Wp[768 + d];
    p2 += av * Wp[1536 + d];
    pm += tv * Wf[d] + vv * Wf[768 + d] + av * Wf[1536 + d];
  }
  s[0][tid] = p0; s[1][tid] = p1; s[2][tid] = p2; s[3][tid] = pm;
  __syncthreads();
  for (int o = 128; o > 0; o >>= 1) {
    if (tid < o) {
      #pragma unroll
      for (int k = 0; k < 4; ++k) s[k][tid] += s[k][tid + o];
    }
    __syncthreads();
  }
  if (tid == 0) {
    out[b] = Wwf[0] * (s[0][0] + bp[0]) + Wwf[1] * (s[1][0] + bp[1])
           + Wwf[2] * (s[2][0] + bp[2]) + Wwf[3] * (s[3][0] + bf[0]);
  }
}

// ---------------------------------------------------------------------------
extern "C" void kernel_launch(void* const* d_in, const int* in_sizes, int n_in,
                              void* d_out, int out_size, void* d_ws, size_t ws_size,
                              hipStream_t stream) {
  (void)in_sizes; (void)n_in; (void)out_size; (void)ws_size;
  const int T = 13;
  const size_t BD = (size_t)BB * DD;
  const size_t stepElems = (size_t)BB * SS * DD;

  const float* hs[3]   = { (const float*)d_in[0], (const float*)d_in[1], (const float*)d_in[2] };
  const float* init[3] = { (const float*)d_in[3], (const float*)d_in[4], (const float*)d_in[5] };
  const float *Wm1[3], *bm1[3], *Watt[3], *batt[3], *Wm2[3], *bm2[3];
  for (int m = 0; m < 3; ++m) {
    const int base = 6 + 6 * m;
    Wm1[m]  = (const float*)d_in[base + 0];
    bm1[m]  = (const float*)d_in[base + 1];
    Watt[m] = (const float*)d_in[base + 2];
    batt[m] = (const float*)d_in[base + 3];
    Wm2[m]  = (const float*)d_in[base + 4];
    bm2[m]  = (const float*)d_in[base + 5];
  }
  const float* Wp  = (const float*)d_in[24];
  const float* bp  = (const float*)d_in[25];
  const float* Wf  = (const float*)d_in[26];
  const float* bf  = (const float*)d_in[27];
  const float* Wwf = (const float*)d_in[28];

  float* ws = (float*)d_ws;   // 18 x BDF floats (~885 KB)

  // carry init (deterministic per call; d2d async is graph-capture safe)
  for (int m = 0; m < 3; ++m)
    hipMemcpyAsync(ws + (size_t)m * BDF, init[m], BD * sizeof(float),
                   hipMemcpyDeviceToDevice, stream);

  for (int t = 0; t < T; ++t) {
    const unsigned long long tOff = (unsigned long long)t * stepElems;
    ln_stage1_k<<<96, 256, 0, stream>>>(hs[0], hs[1], hs[2], tOff, ws);
    gemm_map1_k<<<144, 32, 0, stream>>>(ws, Wm1[0], Wm1[1], Wm1[2],
                                        bm1[0], bm1[1], bm1[2]);
    ln_stage2_k<<<48, 256, 0, stream>>>(ws);
    gemm_gates_k<<<144, 32, 0, stream>>>(ws, Wm2[0], Wm2[1], Wm2[2],
                                         bm2[0], bm2[1], bm2[2]);
    attn_k<<<48, 512, 0, stream>>>(hs[0], hs[1], hs[2], tOff,
                                   Watt[0], Watt[1], Watt[2],
                                   batt[0], batt[1], batt[2], ws);
  }

  final_k<<<BB, 256, 0, stream>>>(ws, Wp, bp, Wf, bf, Wwf, (float*)d_out);
}
